// MoATransformerInteraction_3685081940649
// MI455X (gfx1250) — matrix operand order repaired and checked by reading, not compiled, not run on hardware
//
#include <hip/hip_runtime.h>
#include <math.h>

typedef __bf16 bf16;
typedef __bf16 v16bf __attribute__((ext_vector_type(16)));
typedef __bf16 v8bf  __attribute__((ext_vector_type(8)));
typedef float  v8f   __attribute__((ext_vector_type(8)));

#define DD 256
#define HH 8
#define HD 32
#define PP 12
#define EE 4
#define FFD 512
#define NB 16
#define AA 128
#define BAN 2048
#define TT 24576
#define LN_EPS 1e-5f
#define QK_SCALE 0.17677669529663687f  // 1/sqrt(32)

static __device__ __forceinline__ bf16 f2bf(float f) { return (bf16)f; }

static __device__ __forceinline__ v16bf ld16(const bf16* p0, const bf16* p1) {
  union { v16bf v; v8bf h[2]; } u;
  u.h[0] = *(const v8bf*)p0;
  u.h[1] = *(const v8bf*)p1;
  return u.v;
}

static __device__ __forceinline__ v8f wmma_bf(v16bf a, v16bf b, v8f c) {
  return __builtin_amdgcn_wmma_f32_16x16x32_bf16(false, a, false, b, (short)0, c,
                                                 false, false);
}

// One 16x16 C tile at column-tile nn of: A(16 x K, row-major, stride lda) x
// Bt (transposed weight, N x K row-major, stride ldb). Per WMMA layouts:
// lane = 16*g + mn; A and B per-lane = one 32B chunk each (k = 16g..16g+15, this code's order).
template <int K>
static __device__ __forceinline__ v8f gemm_tile(const bf16* A, int lda,
                                                const bf16* Bt, int ldb,
                                                int nn, int g, int mn) {
  v8f c = {};
  const bf16* arow = A + (size_t)mn * lda;
  const bf16* brow = Bt + (size_t)(nn + mn) * ldb + 16 * g;
#pragma unroll
  for (int kk = 0; kk < K; kk += 32) {
    v16bf a = ld16(arow + kk + 16 * g, arow + kk + 16 * g + 8);
    v16bf b = ld16(brow + kk, brow + kk + 8);
    c = wmma_bf(a, b, c);
  }
  return c;
}

// ---------------- routing: softmax + top-k gates ----------------
__global__ __launch_bounds__(256) void k_routing(const float* __restrict__ q,
                                                 const float* __restrict__ wg,
                                                 const int* __restrict__ kptr,
                                                 float* __restrict__ gates) {
  int t = blockIdx.x * 256 + threadIdx.x;
  if (t >= TT) return;
  float lg[4] = {0.f, 0.f, 0.f, 0.f};
  const float* qr = q + (size_t)t * DD;
  for (int d = 0; d < DD; ++d) {
    float v = qr[d];
    const float* w = wg + d * EE;
    lg[0] += v * w[0]; lg[1] += v * w[1]; lg[2] += v * w[2]; lg[3] += v * w[3];
  }
  float mx = fmaxf(fmaxf(lg[0], lg[1]), fmaxf(lg[2], lg[3]));
  float p[4], s = 0.f;
  for (int e = 0; e < 4; ++e) { p[e] = __expf(lg[e] - mx); s += p[e]; }
  float inv = 1.f / s;
  for (int e = 0; e < 4; ++e) p[e] *= inv;
  int k = *kptr; k = k < 1 ? 1 : (k > 4 ? 4 : k);
  float g[4] = {0.f, 0.f, 0.f, 0.f};
  bool used[4] = {false, false, false, false};
  for (int i = 0; i < k; ++i) {
    int bi = 0; float bv = -1.f;
    for (int e = 0; e < 4; ++e)
      if (!used[e] && p[e] > bv) { bv = p[e]; bi = e; }
    used[bi] = true; g[bi] = bv;
  }
  float* go = gates + (size_t)t * 4;
  go[0] = g[0]; go[1] = g[1]; go[2] = g[2]; go[3] = g[3];
}

// ---------------- km = bf16(key + key_pos) ----------------
__global__ __launch_bounds__(256) void k_km(const float* __restrict__ key,
                                            const float* __restrict__ key_pos,
                                            bf16* __restrict__ kmb) {
  int i = blockIdx.x * 256 + threadIdx.x;  // 16*128*256 = 524288 total
  kmb[i] = f2bf(key[i] + key_pos[i]);
}

// ---------------- weight transpose + bf16 convert: dst[n*K+k] = src[k*N+n] --
__global__ __launch_bounds__(256) void k_tw(const float* __restrict__ src,
                                            bf16* __restrict__ dst, int K, int N) {
  int i = blockIdx.x * 256 + threadIdx.x;
  if (i >= K * N) return;
  int n = i / K, kk = i - n * K;
  dst[i] = f2bf(src[(size_t)kk * N + n]);
}

// ---------------- per-(e,b) cross K/V projections (WMMA) ----------------
__global__ __launch_bounds__(256) void k_kv(const bf16* __restrict__ kmb,
                                            const bf16* __restrict__ wts,
                                            const float* __restrict__ ca_b,
                                            bf16* __restrict__ Kc,
                                            bf16* __restrict__ Vt) {
  int eb = blockIdx.x;             // e*16 + b
  int e = eb >> 4, b = eb & 15;
  int w = threadIdx.x >> 5, lane = threadIdx.x & 31;
  int g = lane >> 4, mn = lane & 15;
  const bf16* A = kmb + ((size_t)b * AA + 16 * w) * DD;  // keys 16w..16w+15
  const bf16* we = wts + (size_t)e * 786432;
  for (int which = 1; which <= 2; ++which) {  // ca wk, wv
    const bf16* Bt = we + 262144 + which * 65536;
    const float* bias = ca_b + (e * 4 + which) * DD;
    for (int t = 0; t < 16; ++t) {
      int nn = 16 * t;
      v8f c = gemm_tile<256>(A, DD, Bt, DD, nn, g, mn);
      float bv = bias[nn + mn];
      if (which == 1) {
#pragma unroll
        for (int r = 0; r < 8; ++r)
          Kc[((size_t)eb * AA + 16 * w + 8 * g + r) * DD + nn + mn] =
              f2bf(c[r] + bv);
      } else {
        v8bf pk;
#pragma unroll
        for (int r = 0; r < 8; ++r) pk[r] = f2bf(c[r] + bv);
        *(v8bf*)&Vt[((size_t)eb * DD + nn + mn) * AA + 16 * w + 8 * g] = pk;
      }
    }
  }
}

// ---------------- LayerNorm over LDS rows (12 real rows, pads zeroed) ------
static __device__ __forceinline__ void layernorm_lds(float* xf, bf16* xb,
                                                     const float* gvec,
                                                     const float* bvec,
                                                     int w, int lane) {
  int half = lane >> 4, c0 = lane & 15;
  if (w < 6) {
    int row = 2 * w + half;
    float vals[16], s = 0.f, s2 = 0.f;
#pragma unroll
    for (int j = 0; j < 16; ++j) {
      float v = xf[row * DD + c0 + 16 * j];
      vals[j] = v; s += v; s2 += v * v;
    }
    s += __shfl_xor(s, 1, 32); s += __shfl_xor(s, 2, 32);
    s += __shfl_xor(s, 4, 32); s += __shfl_xor(s, 8, 32);
    s2 += __shfl_xor(s2, 1, 32); s2 += __shfl_xor(s2, 2, 32);
    s2 += __shfl_xor(s2, 4, 32); s2 += __shfl_xor(s2, 8, 32);
    float mu = s * (1.f / 256.f);
    float var = s2 * (1.f / 256.f) - mu * mu;
    float inv = rsqrtf(var + LN_EPS);
#pragma unroll
    for (int j = 0; j < 16; ++j) {
      int col = c0 + 16 * j;
      float o = (vals[j] - mu) * inv * gvec[col] + bvec[col];
      xf[row * DD + col] = o;
      xb[row * DD + col] = f2bf(o);
    }
  } else {
    int row = 12 + 2 * (w - 6) + half;
#pragma unroll
    for (int j = 0; j < 16; ++j) {
      int col = c0 + 16 * j;
      xf[row * DD + col] = 0.f;
      xb[row * DD + col] = f2bf(0.f);
    }
  }
}

// ---------------- main: fused MoE decoder layer, 1 WG per BA row -----------
__global__ __launch_bounds__(256) void k_decoder(
    const float* __restrict__ query, const float* __restrict__ query_pos,
    const float* __restrict__ gates, const float* __restrict__ sa_b,
    const float* __restrict__ ca_b, const float* __restrict__ ln_g,
    const float* __restrict__ ln_b, const float* __restrict__ fb1,
    const float* __restrict__ fb2, const bf16* __restrict__ wts,
    const bf16* __restrict__ Kc, const bf16* __restrict__ Vt,
    float* __restrict__ out) {
  __shared__ __align__(32) float s_xf[16 * DD];   // residual stream f32
  __shared__ __align__(32) bf16 s_xb[16 * DD];    // bf16 A operand
  __shared__ __align__(32) bf16 s_qb[16 * DD];    // Q proj; aliased as attn-out
  __shared__ __align__(32) bf16 s_un[16 * FFD];   // {self K, self Vt} | FFN h
  __shared__ __align__(32) bf16 s_pw[8 * 16 * 32];// per-wave P (A operand)
  __shared__ __align__(32) bf16 s_z[16];          // zero B chunk (key pad)

  const int tid = threadIdx.x;
  const int w = tid >> 5, lane = tid & 31;
  const int g = lane >> 4, mn = lane & 15;
  const int ba = blockIdx.x;
  const int b = ba >> 7;

  bf16* kb = s_un;              // 16 x 256 (self K, rows 12..15 masked later)
  bf16* vtb = s_un + 16 * DD;   // 256 x 16 (self V transposed)
  bf16* hb = s_un;              // 16 x 512 (FFN hidden)
  bf16* pw = s_pw + w * (16 * 32);

  if (tid < 16) s_z[tid] = f2bf(0.f);

  for (int e = 0; e < EE; ++e) {
    const bf16* we = wts + (size_t)e * 786432;
    const bf16* wqT_sa = we;
    const bf16* wkT_sa = we + 65536;
    const bf16* wvT_sa = we + 131072;
    const bf16* woT_sa = we + 196608;
    const bf16* wqT_ca = we + 262144;
    const bf16* woT_ca = we + 458752;
    const bf16* w1T = we + 524288;
    const bf16* w2T = we + 655360;

    __syncthreads();
    // Ph0: x = query + query_pos (rows >= 12 zero-padded)
    for (int i = tid; i < 16 * DD; i += 256) {
      int m = i >> 8, d = i & 255;
      float v = 0.f;
      if (m < PP) {
        size_t off = ((size_t)ba * PP + m) * DD + d;
        v = query[off] + query_pos[off];
      }
      s_xf[i] = v;
      s_xb[i] = f2bf(v);
    }
    __syncthreads();

    // Ph1: self Q/K/V projections — 48 N-tiles, 6 per wave
    for (int j = 6 * w; j < 6 * w + 6; ++j) {
      int which = j >> 4;
      int nn = (j & 15) << 4;
      const bf16* Bt = (which == 0) ? wqT_sa : (which == 1 ? wkT_sa : wvT_sa);
      v8f c = gemm_tile<256>(s_xb, DD, Bt, DD, nn, g, mn);
      float bv = sa_b[(e * 4 + which) * DD + nn + mn];
      if (which == 0) {
#pragma unroll
        for (int r = 0; r < 8; ++r)
          s_qb[(8 * g + r) * DD + nn + mn] = f2bf(c[r] + bv);
      } else if (which == 1) {
#pragma unroll
        for (int r = 0; r < 8; ++r)
          kb[(8 * g + r) * DD + nn + mn] = f2bf(c[r] + bv);
      } else {
        v8bf pk;
#pragma unroll
        for (int r = 0; r < 8; ++r) pk[r] = f2bf(c[r] + bv);
        *(v8bf*)&vtb[(nn + mn) * 16 + 8 * g] = pk;
      }
    }
    __syncthreads();

    // Ph2: self-attention, one head per wave (H == #waves == 8)
    {
      int h = w;
      const bf16* qrow = s_qb + mn * DD + h * HD;
      v16bf aq = ld16(qrow + 16 * g, qrow + 16 * g + 8);
      const bf16* krow = kb + mn * DD + h * HD + 16 * g;
      v8f sc = {};
      sc = wmma_bf(aq, ld16(krow, krow + 8), sc);
      float sv[8], p[8];
#pragma unroll
      for (int r = 0; r < 8; ++r)
        sv[r] = (mn < PP) ? sc[r] * QK_SCALE : -1e30f;  // mask pad keys
#pragma unroll
      for (int r = 0; r < 8; ++r) {
        float v = sv[r];
        v = fmaxf(v, __shfl_xor(v, 1, 32));
        v = fmaxf(v, __shfl_xor(v, 2, 32));
        v = fmaxf(v, __shfl_xor(v, 4, 32));
        v = fmaxf(v, __shfl_xor(v, 8, 32));
        p[r] = __expf(sv[r] - v);
      }
#pragma unroll
      for (int r = 0; r < 8; ++r) {
        float v = p[r];
        v += __shfl_xor(v, 1, 32);
        v += __shfl_xor(v, 2, 32);
        v += __shfl_xor(v, 4, 32);
        v += __shfl_xor(v, 8, 32);
        p[r] = p[r] / v;
      }
#pragma unroll
      for (int r = 0; r < 8; ++r) {  // P is 16x32 with cols 16..31 zero
        pw[(8 * g + r) * 32 + mn] = f2bf(p[r]);
        pw[(8 * g + r) * 32 + 16 + mn] = f2bf(0.f);
      }
      const bf16* prow = pw + mn * 32;
      v16bf ap = ld16(prow + 16 * g, prow + 16 * g + 8);
#pragma unroll
      for (int half = 0; half < 2; ++half) {
        const bf16* bp = g ? s_z : (vtb + (h * HD + 16 * half + mn) * 16);
        v8f c = {};
        c = wmma_bf(ap, ld16(bp, bp + 8), c);
#pragma unroll
        for (int r = 0; r < 8; ++r)  // attn-out aliases s_qb (own head cols)
          s_qb[(8 * g + r) * DD + h * HD + 16 * half + mn] = f2bf(c[r]);
      }
    }
    __syncthreads();

    // Ph3: self O-proj + residual, then LN0
#pragma unroll
    for (int t = 0; t < 2; ++t) {
      int nn = (2 * w + t) << 4;
      v8f c = gemm_tile<256>(s_qb, DD, woT_sa, DD, nn, g, mn);
      float bv = sa_b[(e * 4 + 3) * DD + nn + mn];
#pragma unroll
      for (int r = 0; r < 8; ++r)
        s_xf[(8 * g + r) * DD + nn + mn] += c[r] + bv;
    }
    __syncthreads();
    layernorm_lds(s_xf, s_xb, ln_g + (e * 3 + 0) * DD, ln_b + (e * 3 + 0) * DD,
                  w, lane);
    __syncthreads();

    // Ph4: cross Q projection
#pragma unroll
    for (int t = 0; t < 2; ++t) {
      int nn = (2 * w + t) << 4;
      v8f c = gemm_tile<256>(s_xb, DD, wqT_ca, DD, nn, g, mn);
      float bv = ca_b[(e * 4 + 0) * DD + nn + mn];
#pragma unroll
      for (int r = 0; r < 8; ++r)
        s_qb[(8 * g + r) * DD + nn + mn] = f2bf(c[r] + bv);
    }
    __syncthreads();

    // Ph5: cross-attention (flash, 4 chunks of 32 keys), one head per wave
    {
      int h = w;
      const bf16* KcB = Kc + ((size_t)e * NB + b) * AA * DD;
      const bf16* VtB = Vt + ((size_t)e * NB + b) * DD * AA;
      const bf16* qrow = s_qb + mn * DD + h * HD;
      v16bf aq = ld16(qrow + 16 * g, qrow + 16 * g + 8);
      float mrun[8], lrun[8];
      v8f acc0 = {}, acc1 = {};
#pragma unroll
      for (int r = 0; r < 8; ++r) { mrun[r] = -1e30f; lrun[r] = 0.f; }
      for (int c4 = 0; c4 < 4; ++c4) {
        int kc = 32 * c4;
        const bf16* k0 = KcB + (size_t)(kc + mn) * DD + h * HD + 16 * g;
        const bf16* k1 = KcB + (size_t)(kc + 16 + mn) * DD + h * HD + 16 * g;
        v8f s0 = {}, s1 = {};
        s0 = wmma_bf(aq, ld16(k0, k0 + 8), s0);
        s1 = wmma_bf(aq, ld16(k1, k1 + 8), s1);
        float p0[8], p1[8];
#pragma unroll
        for (int r = 0; r < 8; ++r) {
          float a0 = s0[r] * QK_SCALE, a1 = s1[r] * QK_SCALE;
          float v = fmaxf(a0, a1);
          v = fmaxf(v, __shfl_xor(v, 1, 32));
          v = fmaxf(v, __shfl_xor(v, 2, 32));
          v = fmaxf(v, __shfl_xor(v, 4, 32));
          v = fmaxf(v, __shfl_xor(v, 8, 32));
          float mnew = fmaxf(mrun[r], v);
          float e0 = __expf(a0 - mnew), e1 = __expf(a1 - mnew);
          float cs = e0 + e1;
          cs += __shfl_xor(cs, 1, 32);
          cs += __shfl_xor(cs, 2, 32);
          cs += __shfl_xor(cs, 4, 32);
          cs += __shfl_xor(cs, 8, 32);
          float resc = __expf(mrun[r] - mnew);
          lrun[r] = lrun[r] * resc + cs;
          acc0[r] *= resc;
          acc1[r] *= resc;
          mrun[r] = mnew;
          p0[r] = e0;
          p1[r] = e1;
        }
#pragma unroll
        for (int r = 0; r < 8; ++r) {  // stage P chunk to reform A layout
          pw[(8 * g + r) * 32 + mn] = f2bf(p0[r]);
          pw[(8 * g + r) * 32 + 16 + mn] = f2bf(p1[r]);
        }
        const bf16* prow = pw + mn * 32;
        v16bf ap = ld16(prow + 16 * g, prow + 16 * g + 8);
        const bf16* v0 = VtB + (size_t)(h * HD + mn) * AA + kc + 16 * g;
        const bf16* v1 = VtB + (size_t)(h * HD + 16 + mn) * AA + kc + 16 * g;
        acc0 = wmma_bf(ap, ld16(v0, v0 + 8), acc0);
        acc1 = wmma_bf(ap, ld16(v1, v1 + 8), acc1);
      }
#pragma unroll
      for (int r = 0; r < 8; ++r) {
        float inv = 1.f / lrun[r];
        s_qb[(8 * g + r) * DD + h * HD + mn] = f2bf(acc0[r] * inv);
        s_qb[(8 * g + r) * DD + h * HD + 16 + mn] = f2bf(acc1[r] * inv);
      }
    }
    __syncthreads();

    // Ph6: cross O-proj + residual, LN1
#pragma unroll
    for (int t = 0; t < 2; ++t) {
      int nn = (2 * w + t) << 4;
      v8f c = gemm_tile<256>(s_qb, DD, woT_ca, DD, nn, g, mn);
      float bv = ca_b[(e * 4 + 3) * DD + nn + mn];
#pragma unroll
      for (int r = 0; r < 8; ++r)
        s_xf[(8 * g + r) * DD + nn + mn] += c[r] + bv;
    }
    __syncthreads();
    layernorm_lds(s_xf, s_xb, ln_g + (e * 3 + 1) * DD, ln_b + (e * 3 + 1) * DD,
                  w, lane);
    __syncthreads();

    // Ph7: FFN up-proj + relu (32 N-tiles, 4 per wave)
#pragma unroll
    for (int t = 0; t < 4; ++t) {
      int nn = (4 * w + t) << 4;
      v8f c = gemm_tile<256>(s_xb, DD, w1T, DD, nn, g, mn);
      float bv = fb1[e * FFD + nn + mn];
#pragma unroll
      for (int r = 0; r < 8; ++r)
        hb[(8 * g + r) * FFD + nn + mn] = f2bf(fmaxf(c[r] + bv, 0.f));
    }
    __syncthreads();
    // Ph8: FFN down-proj + residual, LN2
#pragma unroll
    for (int t = 0; t < 2; ++t) {
      int nn = (2 * w + t) << 4;
      v8f c = gemm_tile<512>(hb, FFD, w2T, FFD, nn, g, mn);
      float bv = fb2[e * DD + nn + mn];
#pragma unroll
      for (int r = 0; r < 8; ++r)
        s_xf[(8 * g + r) * DD + nn + mn] += c[r] + bv;
    }
    __syncthreads();
    layernorm_lds(s_xf, s_xb, ln_g + (e * 3 + 2) * DD, ln_b + (e * 3 + 2) * DD,
                  w, lane);
    __syncthreads();

    // Ph9: gate-weighted output accumulation (deterministic, WG-owned rows)
    for (int i = tid; i < PP * DD; i += 256) {
      int m = i >> 8, d = i & 255;
      size_t t = (size_t)ba * PP + m;
      float gv = gates[t * 4 + e];
      float v = gv * s_xf[m * DD + d];
      if (e == 0) out[t * DD + d] = v;
      else out[t * DD + d] += v;
    }
  }
}

extern "C" void kernel_launch(void* const* d_in, const int* in_sizes, int n_in,
                              void* d_out, int out_size, void* d_ws,
                              size_t ws_size, hipStream_t stream) {
  (void)in_sizes; (void)n_in; (void)out_size; (void)ws_size;
  const float* query = (const float*)d_in[0];
  const float* key = (const float*)d_in[1];
  const float* query_pos = (const float*)d_in[2];
  const float* key_pos = (const float*)d_in[3];
  const float* w_gate = (const float*)d_in[4];
  const float* sa_w = (const float*)d_in[5];
  const float* sa_b = (const float*)d_in[6];
  const float* ca_w = (const float*)d_in[7];
  const float* ca_b = (const float*)d_in[8];
  const float* ln_g = (const float*)d_in[9];
  const float* ln_b = (const float*)d_in[10];
  const float* ffn_w1 = (const float*)d_in[11];
  const float* ffn_b1 = (const float*)d_in[12];
  const float* ffn_w2 = (const float*)d_in[13];
  const float* ffn_b2 = (const float*)d_in[14];
  const int* kptr = (const int*)d_in[15];
  float* out = (float*)d_out;

  char* ws = (char*)d_ws;
  size_t off = 0;
  float* gates = (float*)(ws + off); off += (size_t)TT * 4 * sizeof(float);
  bf16* kmb = (bf16*)(ws + off);     off += (size_t)BAN * DD * sizeof(bf16);
  bf16* Kc = (bf16*)(ws + off);      off += (size_t)EE * NB * AA * DD * sizeof(bf16);
  bf16* Vt = (bf16*)(ws + off);      off += (size_t)EE * NB * AA * DD * sizeof(bf16);
  bf16* wts = (bf16*)(ws + off);     off += (size_t)EE * 786432 * sizeof(bf16);

  k_routing<<<(TT + 255) / 256, 256, 0, stream>>>(query, w_gate, kptr, gates);
  k_km<<<(BAN * DD) / 256, 256, 0, stream>>>(key, key_pos, kmb);

  for (int e = 0; e < EE; ++e) {
    bf16* we = wts + (size_t)e * 786432;
    for (int j = 0; j < 4; ++j)
      k_tw<<<256, 256, 0, stream>>>(sa_w + (size_t)(e * 4 + j) * 65536,
                                    we + j * 65536, 256, 256);
    for (int j = 0; j < 4; ++j)
      k_tw<<<256, 256, 0, stream>>>(ca_w + (size_t)(e * 4 + j) * 65536,
                                    we + 262144 + j * 65536, 256, 256);
    k_tw<<<512, 256, 0, stream>>>(ffn_w1 + (size_t)e * 131072, we + 524288,
                                  256, 512);
    k_tw<<<512, 256, 0, stream>>>(ffn_w2 + (size_t)e * 131072, we + 655360,
                                  512, 256);
  }

  k_kv<<<EE * NB, 256, 0, stream>>>(kmb, wts, ca_b, Kc, Vt);

  k_decoder<<<BAN, 256, 0, stream>>>(query, query_pos, gates, sa_b, ca_b, ln_g,
                                     ln_b, ffn_b1, ffn_b2, wts, Kc, Vt, out);
}
